// Autoformer_47425028882954
// MI455X (gfx1250) — matrix operand order repaired and checked by reading, not compiled, not run on hardware
//
#include <hip/hip_runtime.h>
#include <math.h>

// ---------------- model constants ----------------
#define BB      16
#define LLEN    1024      // SEQ_LEN == DEC_LEN
#define DDIM    512
#define CINCH   8
#define COUTCH  8
#define PREDLEN 512
#define TK      6         // int(log(1024)) = 6
#define MTOT    (BB*LLEN) // 16384

typedef __attribute__((ext_vector_type(16))) _Float16 v16h;
typedef __attribute__((ext_vector_type(8)))  float    v8f;
typedef __attribute__((ext_vector_type(4)))  _Float16 h4;

// ---------------- WMMA GEMM ----------------
// C[M,N] = A[M,K](f32) * B^T  where B is stored as (N,K):
//   B_IS_F16=true : B already f16 (pre-transposed weight, N,K row-major)
//   B_IS_F16=false: B f32 rows (N,K) -> used for Q*K^T (K-matrix rows)
// bias (len N) optional. Batched via blockIdx.z with element strides sA,sB,sC.
// Tile: BM=128 x BN=64 x BK=32, 128 threads (4 waves), each wave does a
// 32x64 sub-tile = 8 WMMAs per k-step (2 A frags + 4 B frags = 12 LDS b128).
// Requires M%128==0, N%64==0, K%32==0, 16B-aligned rows.
#define GBM 128
#define GBN 64
#define GBK 32
#define GLD 40   // GBK + 8 halves pad (80B rows -> 16B aligned)

template<bool B_IS_F16>
__global__ __launch_bounds__(128)
void gemm_wmma(const float* __restrict__ A, const void* __restrict__ Bsrc,
               const float* __restrict__ bias, float* __restrict__ C,
               int N, int K, long long sA, long long sB, long long sC)
{
  __shared__ _Float16 As[GBM * GLD];   // 10240 B
  __shared__ _Float16 Bs[GBN * GLD];   //  5120 B
  const int tid  = threadIdx.x;
  const int wave = tid >> 5;
  const int lane = tid & 31;
  const int hi   = lane >> 4;
  const int lm   = lane & 15;
  const int m0 = blockIdx.y * GBM;
  const int n0 = blockIdx.x * GBN;
  const float*    Af = A + (size_t)blockIdx.z * (size_t)sA;
  const float*    Bf = (const float*)Bsrc + (size_t)blockIdx.z * (size_t)sB;
  const _Float16* Bh = (const _Float16*)Bsrc + (size_t)blockIdx.z * (size_t)sB;
  float* Cf = C + (size_t)blockIdx.z * (size_t)sC;

  v8f acc[2][4] = {};

  for (int k0 = 0; k0 < K; k0 += GBK) {
    // A tile: 128x32 f32 -> f16 LDS (row-major, padded). 1024 float4 chunks.
    #pragma unroll
    for (int i = 0; i < 8; ++i) {
      int idx = tid * 8 + i;
      int r = idx >> 3, c4 = idx & 7;
      float4 v = *(const float4*)(Af + (size_t)(m0 + r) * K + (k0 + c4 * 4));
      h4 h; h[0] = (_Float16)v.x; h[1] = (_Float16)v.y; h[2] = (_Float16)v.z; h[3] = (_Float16)v.w;
      *(h4*)(&As[r * GLD + c4 * 4]) = h;
    }
    // B tile: 64 rows of (N,K) source -> Bs[n][k]
    if (B_IS_F16) {
      #pragma unroll
      for (int i = 0; i < 2; ++i) {
        int idx = tid * 2 + i;         // 256 uint4 chunks (8 halves)
        int r = idx >> 2, c8 = idx & 3;
        uint4 v = *(const uint4*)(Bh + (size_t)(n0 + r) * K + (k0 + c8 * 8));
        *(uint4*)(&Bs[r * GLD + c8 * 8]) = v;
      }
    } else {
      #pragma unroll
      for (int i = 0; i < 4; ++i) {
        int idx = tid * 4 + i;         // 512 float4 chunks
        int r = idx >> 3, c4 = idx & 7;
        float4 v = *(const float4*)(Bf + (size_t)(n0 + r) * K + (k0 + c4 * 4));
        h4 h; h[0] = (_Float16)v.x; h[1] = (_Float16)v.y; h[2] = (_Float16)v.z; h[3] = (_Float16)v.w;
        *(h4*)(&Bs[r * GLD + c4 * 4]) = h;
      }
    }
    __syncthreads();

    // A fragments 16x32: element j (j<8): k = hi*8 + j ; (j>=8): k = 16 + hi*8 + (j-8)
    v16h a[2];
    #pragma unroll
    for (int im = 0; im < 2; ++im) {
      const _Float16* ap = &As[(wave * 32 + im * 16 + lm) * GLD + hi * 8];
      #pragma unroll
      for (int j = 0; j < 8; ++j) { a[im][j] = ap[j]; a[im][8 + j] = ap[16 + j]; }
    }
    // B fragments: element j (j<8): k = hi*8 + j ; (j>=8): k = 16 + hi*8 + (j-8) at column n = lm (Bs is (n,k))
    #pragma unroll
    for (int jn = 0; jn < 4; ++jn) {
      v16h bfrag;
      const _Float16* bp = &Bs[(jn * 16 + lm) * GLD + hi * 8];
      #pragma unroll
      for (int j = 0; j < 8; ++j) { bfrag[j] = bp[j]; bfrag[8 + j] = bp[16 + j]; }
      #pragma unroll
      for (int im = 0; im < 2; ++im)
        acc[im][jn] = __builtin_amdgcn_wmma_f32_16x16x32_f16(
            false, a[im], false, bfrag, (short)0, acc[im][jn], false, false);
    }
    __syncthreads();
  }
  // C/D layout: VGPR r, lanes0-15: m=r; lanes16-31: m=8+r; n=lane&15
  #pragma unroll
  for (int jn = 0; jn < 4; ++jn) {
    int col = n0 + jn * 16 + lm;
    float bv = bias ? bias[col] : 0.0f;
    #pragma unroll
    for (int im = 0; im < 2; ++im) {
      #pragma unroll
      for (int r = 0; r < 8; ++r) {
        int row = m0 + wave * 32 + im * 16 + hi * 8 + r;
        Cf[(size_t)row * N + col] = acc[im][jn][r] + bv;
      }
    }
  }
}

// ---------------- sinusoid tables (PE 1024x512, HOUR 24x512) ----------------
__global__ void tables_kernel(float* __restrict__ PEt, float* __restrict__ HOURt)
{
  int idx = blockIdx.x * blockDim.x + threadIdx.x;
  const int total = 1048 * DDIM;
  if (idx >= total) return;
  int row = idx >> 9;
  int i   = idx & (DDIM - 1);
  double pos = (row < LLEN) ? (double)row : (double)(row - LLEN);
  double expo = (double)(2 * (i / 2)) / (double)DDIM;
  double ang = pos / pow(10000.0, expo);
  float v = (i & 1) ? (float)cos(ang) : (float)sin(ang);
  if (row < LLEN) PEt[idx] = v;
  else            HOURt[(row - LLEN) * DDIM + i] = v;
}

// ---------------- weight f32(K,N) -> f16(N,K) ----------------
__global__ void w2h_kernel(const float* __restrict__ W, _Float16* __restrict__ Wt,
                           int K, int N, int total)
{
  int idx = blockIdx.x * blockDim.x + threadIdx.x;
  if (idx >= total) return;
  int n = idx / K, k = idx - n * K;
  Wt[idx] = (_Float16)W[(size_t)k * N + n];
}

// ---------------- token conv embed + PE + hour ----------------
__global__ void embed_kernel(const float* __restrict__ x, const int* __restrict__ mark,
                             const float* __restrict__ tok, const float* __restrict__ PEt,
                             const float* __restrict__ HOURt, float* __restrict__ out,
                             int total)
{
  int idx = blockIdx.x * blockDim.x + threadIdx.x;
  if (idx >= total) return;
  int d = idx & (DDIM - 1);
  int t = (idx >> 9) & (LLEN - 1);
  int b = idx >> 19;
  float acc = 0.f;
  #pragma unroll
  for (int k = 0; k < 3; ++k) {
    int tt = t - 1 + k; tt = (tt + LLEN) & (LLEN - 1);
    const float* xr = x + ((size_t)(b * LLEN + tt)) * CINCH;
    #pragma unroll
    for (int c = 0; c < CINCH; ++c)
      acc += xr[c] * tok[((size_t)(k * CINCH + c)) * DDIM + d];
  }
  int hr = mark[((size_t)(b * LLEN + t)) * 3 + 2];
  out[idx] = acc + PEt[t * DDIM + d] + HOURt[hr * DDIM + d];
}

// ---------------- series decomposition (25-tap edge-padded mean) ----------------
__global__ void decomp_kernel(const float* __restrict__ x, float* __restrict__ seas,
                              float* __restrict__ trend, int accum, int D, int total)
{
  int idx = blockIdx.x * blockDim.x + threadIdx.x;
  if (idx >= total) return;
  int d = idx % D;
  int t = (idx / D) % LLEN;
  int b = idx / (D * LLEN);
  float s = 0.f;
  #pragma unroll
  for (int j = -12; j <= 12; ++j) {
    int tt = t + j; tt = tt < 0 ? 0 : (tt >= LLEN ? LLEN - 1 : tt);
    s += x[((size_t)(b * LLEN + tt)) * D + d];
  }
  float tr = s * (1.0f / 25.0f);
  if (trend) { if (accum) trend[idx] += tr; else trend[idx] = tr; }
  if (seas) seas[idx] = x[idx] - tr;
}

// ---------------- elementwise ----------------
__global__ void add_kernel(const float* __restrict__ a, const float* __restrict__ b,
                           float* __restrict__ o, int n)
{
  int i = blockIdx.x * blockDim.x + threadIdx.x;
  if (i < n) o[i] = a[i] + b[i];
}
__global__ void gelu_kernel(float* __restrict__ x, int n)
{
  int i = blockIdx.x * blockDim.x + threadIdx.x;
  if (i < n) { float v = x[i]; x[i] = 0.5f * v * (1.0f + erff(v * 0.70710678118654752f)); }
}

// ---------------- layernorm (per-row) ----------------
__global__ __launch_bounds__(256)
void ln_rows_kernel(const float* __restrict__ x, const float* __restrict__ g,
                    const float* __restrict__ bb, float* __restrict__ out)
{
  int row = blockIdx.x;
  const float* xr = x + (size_t)row * DDIM;
  __shared__ float s1[256], s2[256];
  float a = 0.f, q = 0.f;
  for (int d = threadIdx.x; d < DDIM; d += 256) { float v = xr[d]; a += v; q += v * v; }
  s1[threadIdx.x] = a; s2[threadIdx.x] = q; __syncthreads();
  for (int st = 128; st > 0; st >>= 1) {
    if (threadIdx.x < st) { s1[threadIdx.x] += s1[threadIdx.x + st]; s2[threadIdx.x] += s2[threadIdx.x + st]; }
    __syncthreads();
  }
  float mu  = s1[0] * (1.0f / DDIM);
  float var = s2[0] * (1.0f / DDIM) - mu * mu;
  float inv = rsqrtf(var + 1e-5f);
  for (int d = threadIdx.x; d < DDIM; d += 256)
    out[(size_t)row * DDIM + d] = (xr[d] - mu) * inv * g[d] + bb[d];
}
__global__ void colmean_kernel(const float* __restrict__ x, float* __restrict__ cm, int total)
{
  int idx = blockIdx.x * blockDim.x + threadIdx.x;
  if (idx >= total) return;
  int b = idx / DDIM, d = idx - b * DDIM;
  float s = 0.f;
  for (int t = 0; t < LLEN; ++t) s += x[((size_t)(b * LLEN + t)) * DDIM + d];
  cm[idx] = s * (1.0f / LLEN);
}
__global__ void submean_kernel(const float* __restrict__ x, const float* __restrict__ cm,
                               float* __restrict__ o, int total)
{
  int idx = blockIdx.x * blockDim.x + threadIdx.x;
  if (idx >= total) return;
  int d = idx & (DDIM - 1);
  int b = idx >> 19;
  o[idx] = x[idx] - cm[b * DDIM + d];
}

// ---------------- autocorr pieces ----------------
// mean_corr[b,tau] = (1/D) * sum_s S[b, (s+tau)%L, s]
__global__ __launch_bounds__(256)
void diag_kernel(const float* __restrict__ S, float* __restrict__ mc, float invD)
{
  int tau = blockIdx.x, b = blockIdx.y;
  const float* Sb = S + (size_t)b * LLEN * LLEN;
  float s = 0.f;
  for (int t = threadIdx.x; t < LLEN; t += 256) {
    int row = t + tau; if (row >= LLEN) row -= LLEN;
    s += Sb[(size_t)row * LLEN + t];
  }
  __shared__ float red[256];
  red[threadIdx.x] = s; __syncthreads();
  for (int st = 128; st > 0; st >>= 1) {
    if (threadIdx.x < st) red[threadIdx.x] += red[threadIdx.x + st];
    __syncthreads();
  }
  if (threadIdx.x == 0) mc[b * LLEN + tau] = red[0] * invD;
}
__global__ __launch_bounds__(256)
void topk_kernel(const float* __restrict__ mc, float* __restrict__ wts, int* __restrict__ del)
{
  int b = blockIdx.x;
  __shared__ float vals[LLEN];
  __shared__ float rv[256];
  __shared__ int   ri[256];
  __shared__ float tv[TK];
  __shared__ int   ti[TK];
  for (int i = threadIdx.x; i < LLEN; i += 256) vals[i] = mc[b * LLEN + i];
  __syncthreads();
  for (int kk = 0; kk < TK; ++kk) {
    float bv = -3.4e38f; int bi = 0x7fffffff;
    for (int i = threadIdx.x; i < LLEN; i += 256) {
      float v = vals[i];
      if (v > bv || (v == bv && i < bi)) { bv = v; bi = i; }
    }
    rv[threadIdx.x] = bv; ri[threadIdx.x] = bi; __syncthreads();
    for (int st = 128; st > 0; st >>= 1) {
      if (threadIdx.x < st) {
        float v = rv[threadIdx.x + st]; int i2 = ri[threadIdx.x + st];
        if (v > rv[threadIdx.x] || (v == rv[threadIdx.x] && i2 < ri[threadIdx.x])) {
          rv[threadIdx.x] = v; ri[threadIdx.x] = i2;
        }
      }
      __syncthreads();
    }
    if (threadIdx.x == 0) { tv[kk] = rv[0]; ti[kk] = ri[0]; vals[ri[0]] = -3.4e38f; }
    __syncthreads();
  }
  if (threadIdx.x == 0) {
    float m = tv[0], sum = 0.f, e[TK];
    for (int k = 0; k < TK; ++k) { e[k] = expf(tv[k] - m); sum += e[k]; }
    for (int k = 0; k < TK; ++k) { wts[b * TK + k] = e[k] / sum; del[b * TK + k] = ti[k]; }
  }
}
// out[b,l,d] = sum_k w[b,k] * v[b,(l+delay[b,k])%L, d]
__global__ void gather_kernel(const float* __restrict__ v, const float* __restrict__ wts,
                              const int* __restrict__ del, float* __restrict__ o, int total)
{
  int idx = blockIdx.x * blockDim.x + threadIdx.x;
  if (idx >= total) return;
  int d = idx & (DDIM - 1);
  int t = (idx >> 9) & (LLEN - 1);
  int b = idx >> 19;
  float acc = 0.f;
  #pragma unroll
  for (int k = 0; k < TK; ++k) {
    int row = t + del[b * TK + k]; if (row >= LLEN) row -= LLEN;
    acc += wts[b * TK + k] * v[((size_t)(b * LLEN + row)) * DDIM + d];
  }
  o[idx] = acc;
}

// ---------------- trend path ----------------
__global__ void meancols_kernel(const float* __restrict__ x, float* __restrict__ em, int total)
{
  int idx = blockIdx.x * blockDim.x + threadIdx.x;
  if (idx >= total) return;
  int b = idx / CINCH, c = idx - b * CINCH;
  float s = 0.f;
  for (int t = 0; t < LLEN; ++t) s += x[((size_t)(b * LLEN + t)) * CINCH + c];
  em[idx] = s * (1.0f / LLEN);
}
__global__ void trend_build_kernel(const float* __restrict__ ti, const float* __restrict__ em,
                                   float* __restrict__ tr, int total)
{
  int idx = blockIdx.x * blockDim.x + threadIdx.x;
  if (idx >= total) return;
  int c = idx % COUTCH;
  int t = (idx / COUTCH) % LLEN;
  int b = idx / (COUTCH * LLEN);
  tr[idx] = (t < LLEN - PREDLEN) ? ti[((size_t)(b * LLEN + t + PREDLEN)) * COUTCH + c]
                                 : em[b * CINCH + c];
}
// trend[b,t,c] += sum_{k,d} tsum[b,(t-1+k)%L,d] * tw[k,d,c]
__global__ void trend_conv_kernel(const float* __restrict__ ts, const float* __restrict__ tw,
                                  float* __restrict__ tr, int total)
{
  int idx = blockIdx.x * blockDim.x + threadIdx.x;
  if (idx >= total) return;
  int c = idx % COUTCH;
  int t = (idx / COUTCH) % LLEN;
  int b = idx / (COUTCH * LLEN);
  float acc = 0.f;
  for (int k = 0; k < 3; ++k) {
    int tt = t - 1 + k; tt = (tt + LLEN) & (LLEN - 1);
    const float* xr = ts + ((size_t)(b * LLEN + tt)) * DDIM;
    const float* wr = tw + (size_t)k * DDIM * COUTCH + c;
    for (int d = 0; d < DDIM; ++d) acc += xr[d] * wr[(size_t)d * COUTCH];
  }
  tr[idx] += acc;
}
// small projection D->C with bias
__global__ void proj_kernel(const float* __restrict__ x, const float* __restrict__ w,
                            const float* __restrict__ bias, float* __restrict__ o, int total)
{
  int idx = blockIdx.x * blockDim.x + threadIdx.x;
  if (idx >= total) return;
  int c = idx % COUTCH, rc = idx / COUTCH;
  const float* xr = x + (size_t)rc * DDIM;
  float acc = bias[c];
  for (int d = 0; d < DDIM; ++d) acc += xr[d] * w[(size_t)d * COUTCH + c];
  o[idx] = acc;
}
// final 8x8 projection on (trend+seasonal), write sliced outputs
__global__ void final_kernel(const float* __restrict__ tr, const float* __restrict__ se,
                             const float* __restrict__ fw, const float* __restrict__ fb,
                             float* __restrict__ out, int total)
{
  int idx = blockIdx.x * blockDim.x + threadIdx.x;
  if (idx >= total) return;
  int co = idx % COUTCH;
  int to = (idx / COUTCH) % PREDLEN;
  int b  = idx / (COUTCH * PREDLEN);
  int t  = (LLEN - PREDLEN) + to;
  const float* trr = tr + ((size_t)(b * LLEN + t)) * COUTCH;
  const float* ser = se + ((size_t)(b * LLEN + t)) * COUTCH;
  float acc = fb[co];
  #pragma unroll
  for (int ci = 0; ci < COUTCH; ++ci) acc += (trr[ci] + ser[ci]) * fw[ci * COUTCH + co];
  const int part1 = BB * PREDLEN * (COUTCH - 1);
  if (co < COUTCH - 1) out[((size_t)(b * PREDLEN + to)) * (COUTCH - 1) + co] = acc;
  else                 out[part1 + b * PREDLEN + to] = acc;
}

// ======================== host ========================
struct LinP { const float* w; const float* b; };
struct AttnP { LinP q, k, v, o; };

extern "C" void kernel_launch(void* const* d_in, const int* in_sizes, int n_in,
                              void* d_out, int out_size, void* d_ws, size_t ws_size,
                              hipStream_t stream)
{
  auto F = [&](int i) { return (const float*)d_in[i]; };
  // Detect pytree flatten order: JAX-style sorted keys => leaf0 = dec cross k.b (512)
  //                              insertion order       => leaf0 = enc_tok (12288)
  const bool sorted = (n_in > 0 && in_sizes[0] == 512);

  AttnP encA[3]; const float *encW1[3], *encW2[3];
  AttnP decS[2], decC[2]; const float *decW1[2], *decW2[2], *decTW[2];
  const float *enc_tok, *dec_tok, *enc_g, *enc_b, *dec_g, *dec_b, *dp_w, *dp_b, *fp_w, *fp_b;
  const float *x_enc, *x_dec; const int *mk_enc, *mk_dec;

  if (sorted) {
    for (int l = 0; l < 2; ++l) {
      int o = l * 19;
      decC[l] = { {F(o+5),F(o+4)}, {F(o+1),F(o+0)}, {F(o+7),F(o+6)}, {F(o+3),F(o+2)} };
      decS[l] = { {F(o+13),F(o+12)}, {F(o+9),F(o+8)}, {F(o+15),F(o+14)}, {F(o+11),F(o+10)} };
      decTW[l] = F(o+16); decW1[l] = F(o+17); decW2[l] = F(o+18);
    }
    dec_b = F(38); dec_g = F(39); dp_b = F(40); dp_w = F(41); dec_tok = F(42);
    for (int l = 0; l < 3; ++l) {
      int o = 43 + l * 10;
      encA[l] = { {F(o+5),F(o+4)}, {F(o+1),F(o+0)}, {F(o+7),F(o+6)}, {F(o+3),F(o+2)} };
      encW1[l] = F(o+8); encW2[l] = F(o+9);
    }
    enc_b = F(73); enc_g = F(74); enc_tok = F(75); fp_b = F(76); fp_w = F(77);
    x_dec = F(78); x_enc = F(79);
    mk_dec = (const int*)d_in[80]; mk_enc = (const int*)d_in[81];
  } else {
    enc_tok = F(0); dec_tok = F(1);
    for (int l = 0; l < 3; ++l) {
      int o = 2 + l * 10;
      encA[l] = { {F(o+0),F(o+1)}, {F(o+2),F(o+3)}, {F(o+4),F(o+5)}, {F(o+6),F(o+7)} };
      encW1[l] = F(o+8); encW2[l] = F(o+9);
    }
    for (int l = 0; l < 2; ++l) {
      int o = 32 + l * 19;
      decS[l] = { {F(o+0),F(o+1)}, {F(o+2),F(o+3)}, {F(o+4),F(o+5)}, {F(o+6),F(o+7)} };
      decC[l] = { {F(o+8),F(o+9)}, {F(o+10),F(o+11)}, {F(o+12),F(o+13)}, {F(o+14),F(o+15)} };
      decW1[l] = F(o+16); decW2[l] = F(o+17); decTW[l] = F(o+18);
    }
    enc_g = F(70); enc_b = F(71); dec_g = F(72); dec_b = F(73);
    dp_w = F(74); dp_b = F(75); fp_w = F(76); fp_b = F(77);
    x_enc = F(78); mk_enc = (const int*)d_in[79];
    x_dec = F(80); mk_dec = (const int*)d_in[81];
  }

  // -------- workspace layout --------
  char* wsb = (char*)d_ws; size_t off = 0;
  auto alloc = [&](size_t bytes) -> void* {
    void* p = wsb + off; off = (off + bytes + 255) & ~(size_t)255; return p;
  };
  const size_t ABYTES = (size_t)MTOT * DDIM * 4;           // 32 MB activation
  float*    PEt   = (float*)alloc((size_t)LLEN * DDIM * 4);
  float*    HOURt = (float*)alloc((size_t)24 * DDIM * 4);
  _Float16* WT    = (_Float16*)alloc((size_t)38 * DDIM * DDIM * 2);
  float* X    = (float*)alloc(ABYTES);
  float* A1   = (float*)alloc(ABYTES);
  float* A2   = (float*)alloc(ABYTES);
  float* A3   = (float*)alloc(ABYTES);
  float* ENC  = (float*)alloc(ABYTES);
  float* TSUM = (float*)alloc(ABYTES);
  float* S    = (float*)alloc((size_t)BB * LLEN * LLEN * 4); // 64 MB
  float* MC   = (float*)alloc((size_t)BB * LLEN * 4);
  float* WTS  = (float*)alloc((size_t)BB * TK * 4);
  int*   DEL  = (int*)  alloc((size_t)BB * TK * 4);
  float* TREND= (float*)alloc((size_t)BB * LLEN * COUTCH * 4);
  float* TI   = (float*)alloc((size_t)BB * LLEN * COUTCH * 4);
  float* EM   = (float*)alloc((size_t)BB * CINCH * 4);
  float* CM   = (float*)alloc((size_t)BB * DDIM * 4);
  float* SEAS = (float*)alloc((size_t)BB * LLEN * COUTCH * 4);
  (void)ws_size; (void)out_size;

  const int BLD = MTOT * DDIM;
  auto grid1 = [](int n) { return dim3((unsigned)((n + 255) / 256)); };

  // -------- tables --------
  tables_kernel<<<grid1(1048 * DDIM), 256, 0, stream>>>(PEt, HOURt);

  // -------- weight transpose/convert --------
  int nslot = 0;
  auto T = [&](const float* src) -> const _Float16* {
    _Float16* dst = WT + (size_t)nslot * DDIM * DDIM; ++nslot;
    w2h_kernel<<<grid1(DDIM * DDIM), 256, 0, stream>>>(src, dst, DDIM, DDIM, DDIM * DDIM);
    return dst;
  };
  struct AttnH { const _Float16 *q,*k,*v,*o; const float *bq,*bk,*bv,*bo; };
  AttnH encH[3], decSH[2], decCH[2];
  const _Float16 *encH1[3], *encH2[3], *decH1[2], *decH2[2];
  for (int l = 0; l < 3; ++l) {
    encH[l] = { T(encA[l].q.w), T(encA[l].k.w), T(encA[l].v.w), T(encA[l].o.w),
                encA[l].q.b, encA[l].k.b, encA[l].v.b, encA[l].o.b };
    encH1[l] = T(encW1[l]); encH2[l] = T(encW2[l]);
  }
  for (int l = 0; l < 2; ++l) {
    decSH[l] = { T(decS[l].q.w), T(decS[l].k.w), T(decS[l].v.w), T(decS[l].o.w),
                 decS[l].q.b, decS[l].k.b, decS[l].v.b, decS[l].o.b };
    decCH[l] = { T(decC[l].q.w), T(decC[l].k.w), T(decC[l].v.w), T(decC[l].o.w),
                 decC[l].q.b, decC[l].k.b, decC[l].v.b, decC[l].o.b };
    decH1[l] = T(decW1[l]); decH2[l] = T(decW2[l]);
  }

  // -------- GEMM helpers --------
  auto gemm_nn = [&](const float* A, const _Float16* Wn, const float* bias, float* C) {
    gemm_wmma<true><<<dim3(DDIM / GBN, MTOT / GBM, 1), 128, 0, stream>>>(
        A, Wn, bias, C, DDIM, DDIM, 0, 0, 0);
  };
  auto gemm_qkT = [&](const float* Q, const float* Km) {
    gemm_wmma<false><<<dim3(LLEN / GBN, LLEN / GBM, BB), 128, 0, stream>>>(
        Q, Km, nullptr, S, LLEN, DDIM,
        (long long)LLEN * DDIM, (long long)LLEN * DDIM, (long long)LLEN * LLEN);
  };
  // Full autocorrelation attention block; result (o-projection) lands in A1.
  auto run_attn = [&](const float* xq, const float* xkv, const AttnH& W) {
    gemm_nn(xq,  W.q, W.bq, A1);
    gemm_nn(xkv, W.k, W.bk, A2);
    gemm_nn(xkv, W.v, W.bv, A3);
    gemm_qkT(A1, A2);
    diag_kernel<<<dim3(LLEN, BB), 256, 0, stream>>>(S, MC, 1.0f / (float)DDIM);
    topk_kernel<<<BB, 256, 0, stream>>>(MC, WTS, DEL);
    gather_kernel<<<grid1(BLD), 256, 0, stream>>>(A3, WTS, DEL, A2, BLD);
    gemm_nn(A2, W.o, W.bo, A1);
  };

  // -------- trend init (from raw x_enc) --------
  meancols_kernel<<<grid1(BB * CINCH), 256, 0, stream>>>(x_enc, EM, BB * CINCH);
  decomp_kernel<<<grid1(BB * LLEN * CINCH), 256, 0, stream>>>(
      x_enc, nullptr, TI, 0, CINCH, BB * LLEN * CINCH);
  trend_build_kernel<<<grid1(BB * LLEN * COUTCH), 256, 0, stream>>>(
      TI, EM, TREND, BB * LLEN * COUTCH);

  // -------- encoder --------
  embed_kernel<<<grid1(BLD), 256, 0, stream>>>(x_enc, mk_enc, enc_tok, PEt, HOURt, X, BLD);
  for (int l = 0; l < 3; ++l) {
    run_attn(X, X, encH[l]);                                             // A1 = attn(X)
    add_kernel<<<grid1(BLD), 256, 0, stream>>>(X, A1, A2, BLD);          // A2 = X + attn
    decomp_kernel<<<grid1(BLD), 256, 0, stream>>>(A2, X, nullptr, 0, DDIM, BLD); // X = seasonal
    gemm_nn(X, encH1[l], nullptr, A1);
    gelu_kernel<<<grid1(BLD), 256, 0, stream>>>(A1, BLD);
    gemm_nn(A1, encH2[l], nullptr, A2);
    add_kernel<<<grid1(BLD), 256, 0, stream>>>(X, A2, A3, BLD);
    decomp_kernel<<<grid1(BLD), 256, 0, stream>>>(A3, X, nullptr, 0, DDIM, BLD);
  }
  ln_rows_kernel<<<MTOT, 256, 0, stream>>>(X, enc_g, enc_b, A1);
  colmean_kernel<<<grid1(BB * DDIM), 256, 0, stream>>>(A1, CM, BB * DDIM);
  submean_kernel<<<grid1(BLD), 256, 0, stream>>>(A1, CM, ENC, BLD);      // ENC = enc_out

  // -------- decoder --------
  embed_kernel<<<grid1(BLD), 256, 0, stream>>>(x_dec, mk_dec, dec_tok, PEt, HOURt, X, BLD);
  for (int l = 0; l < 2; ++l) {
    run_attn(X, X, decSH[l]);
    add_kernel<<<grid1(BLD), 256, 0, stream>>>(X, A1, A2, BLD);
    decomp_kernel<<<grid1(BLD), 256, 0, stream>>>(A2, X, TSUM, 0, DDIM, BLD); // TSUM = t1
    run_attn(X, ENC, decCH[l]);
    add_kernel<<<grid1(BLD), 256, 0, stream>>>(X, A1, A2, BLD);
    decomp_kernel<<<grid1(BLD), 256, 0, stream>>>(A2, X, TSUM, 1, DDIM, BLD); // += t2
    gemm_nn(X, decH1[l], nullptr, A1);
    gelu_kernel<<<grid1(BLD), 256, 0, stream>>>(A1, BLD);
    gemm_nn(A1, decH2[l], nullptr, A2);
    add_kernel<<<grid1(BLD), 256, 0, stream>>>(X, A2, A3, BLD);
    decomp_kernel<<<grid1(BLD), 256, 0, stream>>>(A3, X, TSUM, 1, DDIM, BLD); // += t3
    trend_conv_kernel<<<grid1(BB * LLEN * COUTCH), 256, 0, stream>>>(
        TSUM, decTW[l], TREND, BB * LLEN * COUTCH);
  }
  ln_rows_kernel<<<MTOT, 256, 0, stream>>>(X, dec_g, dec_b, A1);
  colmean_kernel<<<grid1(BB * DDIM), 256, 0, stream>>>(A1, CM, BB * DDIM);
  submean_kernel<<<grid1(BLD), 256, 0, stream>>>(A1, CM, A1, BLD);
  proj_kernel<<<grid1(BB * LLEN * COUTCH), 256, 0, stream>>>(A1, dp_w, dp_b, SEAS,
                                                             BB * LLEN * COUTCH);
  final_kernel<<<grid1(BB * PREDLEN * COUTCH), 256, 0, stream>>>(
      TREND, SEAS, fp_w, fp_b, (float*)d_out, BB * PREDLEN * COUTCH);
}
